// QLinear_17918603559229
// MI455X (gfx1250) — matrix operand order repaired and checked by reading, not compiled, not run on hardware
//
#include <hip/hip_runtime.h>

// ---- types -----------------------------------------------------------------
typedef __attribute__((ext_vector_type(16))) __bf16       v16bf;
typedef __attribute__((ext_vector_type(8)))  float        v8f;
typedef __attribute__((ext_vector_type(4)))  unsigned int u32x4;
typedef __attribute__((ext_vector_type(4)))  float        f32x4;
typedef __attribute__((ext_vector_type(4)))  __bf16       bf16x4;

#define IN_F   4096
#define OUT_F  4096
#define TOKENS 8192

#define BM  128
#define BN  128
#define BK  32
#define LDT 48   // LDS row stride (bf16 elems): 96B, 16B-aligned rows
#define NIT (IN_F / BK)

// ---- CDNA5 async-to-LDS helpers (inline asm, §15.18.3 / 08_async_tensor) ---
__device__ __forceinline__ void async_ld_b128(unsigned lds_off, const void* gaddr) {
    // VDST = LDS byte-offset VGPR, VADDR = 64-bit global address, GV mode
    asm volatile("global_load_async_to_lds_b128 %0, %1, off"
                 :: "v"(lds_off), "v"(gaddr) : "memory");
}
__device__ __forceinline__ void wait_async0() {
    asm volatile("s_wait_asynccnt 0x0" ::: "memory");
}
__device__ __forceinline__ unsigned lds_off_of(const void* p) {
    // generic LDS pointer = {shared_base_hi32, byte_offset_lo32}
    return (unsigned)(size_t)p;
}

// ---- kernel 1: materialize w = sum_k kron(a[k], s[k]) as bf16 hi/lo --------
__global__ __launch_bounds__(256) void prep_w(const float* __restrict__ a,
                                              const float* __restrict__ s,
                                              __bf16* __restrict__ whi,
                                              __bf16* __restrict__ wlo) {
    int t  = blockIdx.x * 256 + threadIdx.x;   // 4096*1024 threads, 4 c-elems each
    int o  = t >> 10;
    int cq = (t & 1023) << 2;
    int i  = o >> 10, p = o & 1023;
    int j  = cq >> 10, q = cq & 1023;

    f32x4 acc = {0.f, 0.f, 0.f, 0.f};
#pragma unroll
    for (int k = 0; k < 4; ++k) {
        float ak = a[(k * 4 + i) * 4 + j];
        f32x4 sv = *(const f32x4*)(s + ((size_t)(k * 1024 + p)) * 1024 + q);
        acc += ak * sv;
    }
    bf16x4 hi, lo;
#pragma unroll
    for (int e = 0; e < 4; ++e) {
        __bf16 h = (__bf16)acc[e];
        hi[e] = h;
        lo[e] = (__bf16)(acc[e] - (float)h);
    }
    size_t base = (size_t)o * IN_F + cq;
    *(bf16x4*)(whi + base) = hi;
    *(bf16x4*)(wlo + base) = lo;
}

// ---- kernel 2: y = x @ w^T + bias, bf16x3 split WMMA, double-buffered ------
__global__ __launch_bounds__(256) void kron_gemm(const float*  __restrict__ x,
                                                 const __bf16* __restrict__ whi,
                                                 const __bf16* __restrict__ wlo,
                                                 const float*  __restrict__ bias,
                                                 float*        __restrict__ out) {
    __shared__ __bf16 XsH[2][BM * LDT];
    __shared__ __bf16 XsL[2][BM * LDT];
    __shared__ __bf16 WsH[2][BN * LDT];
    __shared__ __bf16 WsL[2][BN * LDT];

    const int tid  = threadIdx.x;
    const int lane = tid & 31;
    const int wave = tid >> 5;
    const int half = lane >> 4;
    const int l16  = lane & 15;

    const int bm = blockIdx.y * BM;
    const int bn = blockIdx.x * BN;

    const int waveRow = wave >> 1;  // 0..3 -> rows 32*waveRow..+32
    const int waveCol = wave & 1;   // 0..1 -> cols 64*waveCol..+64

    v8f acc[2][4];
    {
        v8f z = {};
#pragma unroll
        for (int tm = 0; tm < 2; ++tm)
#pragma unroll
            for (int tn = 0; tn < 4; ++tn) acc[tm][tn] = z;
    }

    union Frag { v16bf v; u32x4 q[2]; };

    // --- W-tile prefetch via async-to-LDS (bf16 data, direct path) ----------
    auto issue_w = [&](int buf, int k0) {
#pragma unroll
        for (int part = 0; part < 2; ++part) {
            int lin = tid + part * 256;          // 0..511 chunks of 8 bf16
            int n   = lin >> 2;                  // row 0..127
            int ch  = (lin & 3) << 3;            // bf16 col 0,8,16,24
            size_t g = (size_t)(bn + n) * IN_F + k0 + ch;
            async_ld_b128(lds_off_of(&WsH[buf][n * LDT + ch]), whi + g);
            async_ld_b128(lds_off_of(&WsL[buf][n * LDT + ch]), wlo + g);
        }
    };
    // --- X-tile: global f32 -> regs (issue early), split to LDS (late) ------
    auto issue_x = [&](int k0, f32x4* xv) {
#pragma unroll
        for (int part = 0; part < 4; ++part) {
            int lin = tid + part * 256;          // 0..1023
            int r   = lin >> 3;                  // row 0..127
            int ch  = (lin & 7) << 2;            // f32 col
            xv[part] = *(const f32x4*)(x + (size_t)(bm + r) * IN_F + k0 + ch);
        }
    };
    auto store_x = [&](int buf, const f32x4* xv) {
#pragma unroll
        for (int part = 0; part < 4; ++part) {
            int lin = tid + part * 256;
            int r   = lin >> 3;
            int ch  = (lin & 7) << 2;
            bf16x4 h, l;
#pragma unroll
            for (int e = 0; e < 4; ++e) {
                __bf16 hh = (__bf16)xv[part][e];
                h[e] = hh;
                l[e] = (__bf16)(xv[part][e] - (float)hh);
            }
            *(bf16x4*)(&XsH[buf][r * LDT + ch]) = h;
            *(bf16x4*)(&XsL[buf][r * LDT + ch]) = l;
        }
    };

    // --- prologue: tile 0 into buffer 0 -------------------------------------
    {
        f32x4 xv[4];
        issue_w(0, 0);
        issue_x(0, xv);
        store_x(0, xv);
        wait_async0();
        __syncthreads();
    }

    for (int k = 0; k < NIT; ++k) {
        const int p  = k & 1;
        const int q_ = p ^ 1;
        const int k1 = (k + 1) * BK;
        const bool pf = (k + 1) < NIT;

        f32x4 xv[4];
        if (pf) {                 // overlap next-tile loads with this tile's WMMAs
            issue_w(q_, k1);      // async direct-to-LDS, tracked by ASYNCcnt
            issue_x(k1, xv);      // f32 loads into regs, converted after compute
        }

        // --- fragments from buffer p (ISA 7.12.2 16-bit layouts) ---
        v16bf afH[2], afL[2], bfH[4], bfL[4];
#pragma unroll
        for (int tm = 0; tm < 2; ++tm) {
            int row = 32 * waveRow + 16 * tm + l16;
            const __bf16* ph = &XsH[p][row * LDT + 8 * half];
            const __bf16* pl = &XsL[p][row * LDT + 8 * half];
            Frag fh, fl;
            fh.q[0] = *(const u32x4*)(ph);        // K = 8*half .. +7
            fh.q[1] = *(const u32x4*)(ph + 16);   // K = 16+8*half .. +7
            fl.q[0] = *(const u32x4*)(pl);
            fl.q[1] = *(const u32x4*)(pl + 16);
            afH[tm] = fh.v;
            afL[tm] = fl.v;
        }
#pragma unroll
        for (int tn = 0; tn < 4; ++tn) {
            int nrow = 64 * waveCol + 16 * tn + l16;   // one N column per lane
            const __bf16* ph = &WsH[p][nrow * LDT + 8 * half];
            const __bf16* pl = &WsL[p][nrow * LDT + 8 * half];
            Frag fh, fl;
            fh.q[0] = *(const u32x4*)(ph);        // K = 8*half .. +7
            fh.q[1] = *(const u32x4*)(ph + 16);   // K = 16+8*half .. +7
            fl.q[0] = *(const u32x4*)(pl);
            fl.q[1] = *(const u32x4*)(pl + 16);
            bfH[tn] = fh.v;
            bfL[tn] = fl.v;
        }

        // --- bf16x3: hi*hi + hi*lo + lo*hi, f32 accumulate ---
#pragma unroll
        for (int tm = 0; tm < 2; ++tm)
#pragma unroll
            for (int tn = 0; tn < 4; ++tn) {
                acc[tm][tn] = __builtin_amdgcn_wmma_f32_16x16x32_bf16(
                    false, afH[tm], false, bfH[tn], (short)0, acc[tm][tn], false, false);
                acc[tm][tn] = __builtin_amdgcn_wmma_f32_16x16x32_bf16(
                    false, afH[tm], false, bfL[tn], (short)0, acc[tm][tn], false, false);
                acc[tm][tn] = __builtin_amdgcn_wmma_f32_16x16x32_bf16(
                    false, afL[tm], false, bfH[tn], (short)0, acc[tm][tn], false, false);
            }

        if (pf) store_x(q_, xv);  // f32->bf16 hi/lo split into next buffer

        wait_async0();            // W tiles for k+1 landed in LDS
        __syncthreads();          // (also flushes DScnt for the X stores)
    }

    // --- epilogue: C layout (VGPR r -> M = r + 8*half, N = l16) + bias ---
#pragma unroll
    for (int tm = 0; tm < 2; ++tm)
#pragma unroll
        for (int tn = 0; tn < 4; ++tn) {
            int col = bn + 64 * waveCol + 16 * tn + l16;
            float bv = bias[col];
            int rbase = bm + 32 * waveRow + 16 * tm + 8 * half;
#pragma unroll
            for (int r = 0; r < 8; ++r) {
                out[(size_t)(rbase + r) * OUT_F + col] = acc[tm][tn][r] + bv;
            }
        }
}

// ---- launcher --------------------------------------------------------------
extern "C" void kernel_launch(void* const* d_in, const int* in_sizes, int n_in,
                              void* d_out, int out_size, void* d_ws, size_t ws_size,
                              hipStream_t stream) {
    const float* x    = (const float*)d_in[0];
    const float* a    = (const float*)d_in[1];
    const float* s    = (const float*)d_in[2];
    const float* bias = (const float*)d_in[3];
    float* out = (float*)d_out;

    __bf16* whi = (__bf16*)d_ws;                       // 32 MB
    __bf16* wlo = whi + (size_t)OUT_F * IN_F;          // 32 MB

    prep_w<<<(OUT_F * (IN_F / 4)) / 256, 256, 0, stream>>>(a, s, whi, wlo);

    dim3 grid(OUT_F / BN, TOKENS / BM);
    kron_gemm<<<grid, 256, 0, stream>>>(x, whi, wlo, bias, out);
}
